// HybridQNN_64750926954779
// MI455X (gfx1250) — matrix operand order repaired and checked by reading, not compiled, not run on hardware
//
#include <hip/hip_runtime.h>
#include <hip/hip_bf16.h>

#define NQ 10
#define DIM 1024           // 2^NQ
#define NLAYERS 2
#define BATCH 4096
#define QSTRIDE 16         // q rows zero-padded to 16 floats

typedef __attribute__((ext_vector_type(16))) _Float16 v16h;
typedef __attribute__((ext_vector_type(8)))  float    v8f;

// ds_swizzle XOR butterfly: offset = {1'b0, xor[14:10], or[9:5]=0, and[4:0]=0x1F}
template <int XORMASK>
__device__ __forceinline__ float swz_xor(float v) {
    int i = __builtin_bit_cast(int, v);
    i = __builtin_amdgcn_ds_swizzle(i, (XORMASK << 10) | 0x1F);
    return __builtin_bit_cast(float, i);
}

// ---------------------------------------------------------------------------
// Kernel 1: 10-qubit statevector simulation, one block (512 threads) per
// batch sample. State lives in LDS (8KB). Emits q[b][0..15] = <Z_w> (padded).
// Block 0 additionally packs w1 into the per-lane WMMA B-fragment layout.
// ---------------------------------------------------------------------------
__global__ __launch_bounds__(512)
void qsim_kernel(const float* __restrict__ x,      // (BATCH, NQ)
                 const float* __restrict__ wts,    // (NLAYERS, NQ, 3)
                 const float* __restrict__ w1,     // (8, NQ)
                 float* __restrict__ qout,         // (BATCH, QSTRIDE)
                 _Float16* __restrict__ bpack)     // (32 lanes, 16 halves)
{
    __shared__ float sre[DIM];
    __shared__ float sim_[DIM];
    __shared__ float zsum[NQ];

    const int b   = blockIdx.x;
    const int tid = threadIdx.x;            // 0..511

    __builtin_prefetch(&x[b * NQ], 0, 0);   // global_prefetch_b8

    // Pack B = w1^T (32x16 f16, ISA 16-bit B layout) once, in block 0.
    if (b == 0 && tid < 32) {
        const int lane = tid;
        const int col  = lane & 15;
#pragma unroll
        for (int j = 0; j < 16; ++j) {
            const int k = (j < 8) ? (((lane < 16) ? 0 : 8) + j) : (16 + ((lane < 16) ? 0 : 8) + (j - 8));
            const float bv = (col < 8 && k < NQ) ? w1[col * NQ + k] : 0.0f;
            bpack[lane * 16 + j] = (_Float16)bv;
        }
    }

    // |0...0>
    sre[tid]        = 0.0f;  sim_[tid]        = 0.0f;
    sre[tid + 512]  = 0.0f;  sim_[tid + 512]  = 0.0f;
    if (tid == 0) sre[0] = 1.0f;
    if (tid < NQ) zsum[tid] = 0.0f;
    __syncthreads();

    // ---- AngleEmbedding: RY(x[b][w]) on wire w (wire w <-> bit 9-w) ----
    for (int w = 0; w < NQ; ++w) {
        const int   k  = (NQ - 1) - w;
        const float th = 0.5f * x[b * NQ + w];
        float s, c;
        __sincosf(th, &s, &c);
        const int msk = (1 << k) - 1;
        const int i0  = ((tid >> k) << (k + 1)) | (tid & msk);
        const int i1  = i0 | (1 << k);
        const float a0r = sre[i0], a0i = sim_[i0];
        const float a1r = sre[i1], a1i = sim_[i1];
        sre[i0] = c * a0r - s * a1r;  sim_[i0] = c * a0i - s * a1i;
        sre[i1] = s * a0r + c * a1r;  sim_[i1] = s * a0i + c * a1i;
        __syncthreads();
    }

    // ---- StronglyEntanglingLayers ----
    for (int l = 0; l < NLAYERS; ++l) {
        // Rot(phi, theta, omega) on each wire
        for (int w = 0; w < NQ; ++w) {
            const float phi = wts[(l * NQ + w) * 3 + 0];
            const float th  = wts[(l * NQ + w) * 3 + 1];
            const float om  = wts[(l * NQ + w) * 3 + 2];
            float st, ct;  __sincosf(0.5f * th, &st, &ct);
            float epr, epi; { const float a = -0.5f * (phi + om); __sincosf(a, &epi, &epr); }
            float emr, emi; { const float a =  0.5f * (phi - om); __sincosf(a, &emi, &emr); }
            // U = [[ep*ct, -em*st], [conj(em)*st, conj(ep)*ct]]
            const float u00r =  epr * ct, u00i =  epi * ct;
            const float u01r = -emr * st, u01i = -emi * st;
            const float u10r =  emr * st, u10i = -emi * st;
            const float u11r =  epr * ct, u11i = -epi * ct;

            const int k   = (NQ - 1) - w;
            const int msk = (1 << k) - 1;
            const int i0  = ((tid >> k) << (k + 1)) | (tid & msk);
            const int i1  = i0 | (1 << k);
            const float a0r = sre[i0], a0i = sim_[i0];
            const float a1r = sre[i1], a1i = sim_[i1];
            sre[i0]  = u00r * a0r - u00i * a0i + u01r * a1r - u01i * a1i;
            sim_[i0] = u00r * a0i + u00i * a0r + u01r * a1i + u01i * a1r;
            sre[i1]  = u10r * a0r - u10i * a0i + u11r * a1r - u11i * a1i;
            sim_[i1] = u10r * a0i + u10i * a0r + u11r * a1i + u11i * a1r;
            __syncthreads();
        }
        // CNOT ring, range r = (l % (NQ-1)) + 1
        const int rr = (l % (NQ - 1)) + 1;
        for (int w = 0; w < NQ; ++w) {
            const int kc = (NQ - 1) - w;
            const int kt = (NQ - 1) - ((w + rr) % NQ);
            if (tid < 256) {
                const int lo = (kc < kt) ? kc : kt;
                const int hi = (kc < kt) ? kt : kc;
                int t1 = ((tid >> lo) << (lo + 1)) | (tid & ((1 << lo) - 1));
                int t2 = ((t1  >> hi) << (hi + 1)) | (t1  & ((1 << hi) - 1));
                const int iA = t2 | (1 << kc);     // control=1, target=0
                const int iB = iA | (1 << kt);     // control=1, target=1
                const float rA = sre[iA], mA = sim_[iA];
                const float rB = sre[iB], mB = sim_[iB];
                sre[iA] = rB; sim_[iA] = mB;
                sre[iB] = rA; sim_[iB] = mA;
            }
            __syncthreads();
        }
    }

    // ---- probabilities -> per-wire <Z> ----
    float zp[NQ];
#pragma unroll
    for (int w = 0; w < NQ; ++w) zp[w] = 0.0f;
#pragma unroll
    for (int j = 0; j < 2; ++j) {
        const int   i = tid + j * 512;
        const float p = sre[i] * sre[i] + sim_[i] * sim_[i];
#pragma unroll
        for (int w = 0; w < NQ; ++w)
            zp[w] += ((i >> ((NQ - 1) - w)) & 1) ? -p : p;
    }
#pragma unroll
    for (int w = 0; w < NQ; ++w) {
        float v = zp[w];
        v += swz_xor<16>(v);                    // ds_swizzle butterflies
        v += swz_xor<8>(v);
        v += swz_xor<4>(v);
        v += swz_xor<2>(v);
        v += swz_xor<1>(v);
        if ((tid & 31) == 0) atomicAdd(&zsum[w], v);   // ds_add_f32
    }
    __syncthreads();
    if (tid < QSTRIDE) qout[b * QSTRIDE + tid] = (tid < NQ) ? zsum[tid] : 0.0f;
}

// ---------------------------------------------------------------------------
// Kernel 2: MLP head via WMMA. One wave per 16-row tile of q.
//   h = relu(q @ w1^T + b1) via v_wmma_f32_16x16x32_f16 (K padded 10->32,
//   N padded 8->16), then out = sigmoid(h @ w2^T + b2) via ds_swizzle
//   reduction. All operand loads are unconditional vector loads.
// ---------------------------------------------------------------------------
__global__ __launch_bounds__(256)
void mlp_head_kernel(const float* __restrict__ q,        // (BATCH, QSTRIDE)
                     const _Float16* __restrict__ bpack, // (32, 16) halves
                     const float* __restrict__ b1,       // (8,)
                     const float* __restrict__ w2,       // (1, 8)
                     const float* __restrict__ b2,       // (1,)
                     float* __restrict__ out)            // (BATCH,)
{
    const int wave = (blockIdx.x * blockDim.x + threadIdx.x) >> 5;
    const int lane = threadIdx.x & 31;
    const int row0 = wave * 16;                 // 16 batch rows per wave
    const int sub  = lane & 15;                 // column / row-in-half index
    const bool hiHalf = (lane >= 16);

    // --- A: q tile 16x32 f16. Lane<16 carries K=0..7; lane>=16 carries
    //     K=8..15 (10..15 are the zero pad). K>=16 slots are zero. ---
    v16h a;
    {
        const float* base = q + (row0 + sub) * QSTRIDE + (hiHalf ? 8 : 0);
        const float4 lo = *(const float4*)(base + 0);
        const float4 hi = *(const float4*)(base + 4);
        a[0] = (_Float16)lo.x; a[1] = (_Float16)lo.y;
        a[2] = (_Float16)lo.z; a[3] = (_Float16)lo.w;
        a[4] = (_Float16)hi.x; a[5] = (_Float16)hi.y;
        a[6] = (_Float16)hi.z; a[7] = (_Float16)hi.w;
#pragma unroll
        for (int j = 8; j < 16; ++j) a[j] = (_Float16)0.0f;
    }
    // --- B: pre-packed per-lane fragment (single 32B vector load) ---
    const v16h bm = *(const v16h*)(bpack + lane * 16);

    // --- C: broadcast bias b1 along rows ---
    const float bias1 = (sub < 8) ? b1[sub] : 0.0f;
    v8f c;
#pragma unroll
    for (int r = 0; r < 8; ++r) c[r] = bias1;

    // D = A x B + C   -> v_wmma_f32_16x16x32_f16
    v8f d = __builtin_amdgcn_wmma_f32_16x16x32_f16(
        /*neg_a=*/false, a, /*neg_b=*/false, bm,
        /*c_mod=*/(short)0, c, /*reuse_a=*/false, /*reuse_b=*/false);

    // --- second layer: out[m] = sigmoid(sum_j relu(h[m][j]) * w2[j] + b2) ---
    const float w2v   = (sub < 8) ? w2[sub] : 0.0f;
    const float bias2 = b2[0];
#pragma unroll
    for (int r = 0; r < 8; ++r) {
        float v = fmaxf(d[r], 0.0f) * w2v;      // relu then weight
        v += swz_xor<1>(v);                     // reduce 16-lane column group
        v += swz_xor<2>(v);
        v += swz_xor<4>(v);
        v += swz_xor<8>(v);
        if (sub == 0) {
            const int m = hiHalf ? (8 + r) : r;  // D layout: VGPR r -> M=r / 8+r
            const float z = v + bias2;
            out[row0 + m] = 1.0f / (1.0f + __expf(-z));
        }
    }
}

// ---------------------------------------------------------------------------
extern "C" void kernel_launch(void* const* d_in, const int* in_sizes, int n_in,
                              void* d_out, int out_size, void* d_ws, size_t ws_size,
                              hipStream_t stream) {
    const float* x   = (const float*)d_in[0];   // (4096, 10)
    const float* wts = (const float*)d_in[1];   // (2, 10, 3)
    const float* w1  = (const float*)d_in[2];   // (8, 10)
    const float* b1  = (const float*)d_in[3];   // (8,)
    const float* w2  = (const float*)d_in[4];   // (1, 8)
    const float* b2  = (const float*)d_in[5];   // (1,)
    float*       out = (float*)d_out;           // (4096, 1)

    // Workspace layout: q (4096 x 16 floats) then packed B fragment (1KB).
    float*    q     = (float*)d_ws;
    _Float16* bpack = (_Float16*)((char*)d_ws + BATCH * QSTRIDE * sizeof(float));

    qsim_kernel<<<BATCH, 512, 0, stream>>>(x, wts, w1, q, bpack);

    // 4096 rows / 16 per wave = 256 waves; 8 waves (256 thr) per block.
    mlp_head_kernel<<<BATCH / 16 / 8, 256, 0, stream>>>(q, bpack, b1, w2, b2, out);
}
